// MambaBlock_53824530153926
// MI455X (gfx1250) — matrix operand order repaired and checked by reading, not compiled, not run on hardware
//
#include <hip/hip_runtime.h>
#include <hip/hip_bf16.h>

#define D_MODEL 1024
#define D_INNER 2048
#define D_STATE 16
#define D_CONV  4
#define DT_RANK 128
#define BATCH   2
#define SEQLEN  1024
#define NROWS   (BATCH * SEQLEN)          // 2048 rows (M dim of all GEMMs)
#define XPJ_N   (DT_RANK + 2 * D_STATE)   // 160

typedef __attribute__((ext_vector_type(16))) __bf16 v16bf;
typedef __attribute__((ext_vector_type(8)))  float  v8f;

// ---------------------------------------------------------------- helpers
__device__ __forceinline__ __bf16 f2bf(float f) {
    union { float f; unsigned u; } v; v.f = f;
    unsigned r = v.u + 0x7fffu + ((v.u >> 16) & 1u);   // round-nearest-even
    union { unsigned short s; __bf16 b; } o; o.s = (unsigned short)(r >> 16);
    return o.b;
}
__device__ __forceinline__ float softplus_f(float x) {
    return (x > 20.f) ? x : log1pf(__expf(x));
}
__device__ __forceinline__ float silu_f(float x) {
    return x / (1.f + __expf(-x));
}

// ---------------------------------------------------------------- layernorm + bf16 cast
__global__ void __launch_bounds__(256)
ln_cast_kernel(const float* __restrict__ x, const float* __restrict__ g,
               const float* __restrict__ bt, __bf16* __restrict__ out) {
    __shared__ float ssum[8], ssum2[8];
    int row = blockIdx.x;
    const float* xr = x + (size_t)row * D_MODEL;
    float s = 0.f, s2 = 0.f;
    for (int i = threadIdx.x; i < D_MODEL; i += 256) { float v = xr[i]; s += v; s2 += v * v; }
    for (int o = 16; o; o >>= 1) { s += __shfl_xor(s, o, 32); s2 += __shfl_xor(s2, o, 32); }
    int wv = threadIdx.x >> 5, ln = threadIdx.x & 31;
    if (ln == 0) { ssum[wv] = s; ssum2[wv] = s2; }
    __syncthreads();
    if (wv == 0) {
        s  = (ln < 8) ? ssum[ln]  : 0.f;
        s2 = (ln < 8) ? ssum2[ln] : 0.f;
        for (int o = 4; o; o >>= 1) { s += __shfl_xor(s, o, 32); s2 += __shfl_xor(s2, o, 32); }
        if (ln == 0) { ssum[0] = s; ssum2[0] = s2; }
    }
    __syncthreads();
    float mu  = ssum[0] / D_MODEL;
    float var = ssum2[0] / D_MODEL - mu * mu;
    float inv = rsqrtf(var + 1e-5f);
    __bf16* orow = out + (size_t)row * D_MODEL;
    for (int i = threadIdx.x; i < D_MODEL; i += 256)
        orow[i] = f2bf((xr[i] - mu) * inv * g[i] + bt[i]);
}

// ---------------------------------------------------------------- f32 -> bf16 transpose-cast
// src: K x N (f32, row-major)  ->  dst: N x K (bf16, row-major)  == W^T
__global__ void __launch_bounds__(256)
transpose_cast_bf16_kernel(const float* __restrict__ src, __bf16* __restrict__ dst,
                           int K, int N) {
    int i = blockIdx.x * 256 + threadIdx.x;     // coalesced read over K*N
    if (i >= K * N) return;
    int k = i / N, n = i % N;
    dst[(size_t)n * K + k] = f2bf(src[i]);
}

// ---------------------------------------------------------------- WMMA bf16 GEMM
// C[M,N](f32) = A[M,K](bf16, lda) @ W[K,N]  where Wt = W^T is N x K (bf16, row stride K)
// mode 0: plain   mode 1: softplus(C + bias[n])   mode 2: C + resid[m*ldr+n]
// Cbf != nullptr -> also emit bf16 copy of the result (same ldc).
#define TM 64
#define TN 128
#define TK 32
__global__ void __launch_bounds__(128)
gemm_bf16_wmma(const __bf16* __restrict__ A, int lda,
               const __bf16* __restrict__ Wt,            // N x K, row stride K
               float* __restrict__ C, int ldc,
               __bf16* __restrict__ Cbf,
               const float* __restrict__ bias,
               const float* __restrict__ resid, int ldr,
               int M, int N, int K, int mode) {
    __shared__ __bf16 As [TM][TK];   // A tile  (m, k)
    __shared__ __bf16 Bst[TN][TK];   // B tile, already transposed: (n, k)

    const int tid  = threadIdx.x;          // 128 threads = 4 waves
    const int wave = tid >> 5;
    const int lane = tid & 31;
    const int m0 = blockIdx.y * TM;
    const int n0 = blockIdx.x * TN;

    v8f acc[8] = {};                        // 8 N-tiles of 16x16 f32 per wave

    const int ar = tid >> 1;                // A-load: row in tile (0..63)
    const int as = (tid & 1) * 16;          // A-load: 16-elem segment

    for (int k0 = 0; k0 < K; k0 += TK) {
        // ---- stage A tile (64x32 bf16): 2x b128 per thread ----
        {
            const __bf16* src = A + (size_t)(m0 + ar) * lda + k0 + as;
            uint4 u0 = *(const uint4*)src;
            uint4 u1 = *(const uint4*)(src + 8);
            *(uint4*)&As[ar][as] = u0;
            *(uint4*)&As[ar][as + 8] = u1;
            if (k0 + TK < K)
                __builtin_prefetch(src + TK, 0, 1);   // global_prefetch_b8
        }
        // ---- stage B tile (128x32 bf16, (n,k) layout): 4x b128 per thread ----
        {
            uint4 u0, u1, u2, u3;
            if (n0 + tid < N) {
                const __bf16* src = Wt + (size_t)(n0 + tid) * K + k0;
                u0 = *(const uint4*)src;
                u1 = *(const uint4*)(src + 8);
                u2 = *(const uint4*)(src + 16);
                u3 = *(const uint4*)(src + 24);
                if (k0 + TK < K)
                    __builtin_prefetch(src + TK, 0, 1);
            } else {
                u0 = make_uint4(0, 0, 0, 0);
                u1 = make_uint4(0, 0, 0, 0);
                u2 = make_uint4(0, 0, 0, 0);
                u3 = make_uint4(0, 0, 0, 0);
            }
            *(uint4*)&Bst[tid][0] = u0;
            *(uint4*)&Bst[tid][8] = u1;
            *(uint4*)&Bst[tid][16] = u2;
            *(uint4*)&Bst[tid][24] = u3;
        }
        __syncthreads();

        // ---- A fragment: lane = M row; k(e) = 8*half + e (e<8) / 8 + 8*half + e (e>=8)
        const int m  = wave * 16 + (lane & 15);
        const int hh = (lane >> 4) * 8;
        union { uint4 u[2]; v16bf v; } ta;
        ta.u[0] = *(const uint4*)&As[m][hh];
        ta.u[1] = *(const uint4*)&As[m][16 + hh];
        v16bf af = ta.v;

        const int nl = lane & 15;
        const int kb = (lane >> 4) * 8;    // B fragment: k(e) = 8*half + e (e<8) / 8 + 8*half + e (e>=8)
#pragma unroll
        for (int nt = 0; nt < 8; ++nt) {
            union { uint4 u[2]; v16bf v; } tb;
            tb.u[0] = *(const uint4*)&Bst[nt * 16 + nl][kb];
            tb.u[1] = *(const uint4*)&Bst[nt * 16 + nl][kb + 16];
            acc[nt] = __builtin_amdgcn_wmma_f32_16x16x32_bf16(
                false, af, false, tb.v, (short)0, acc[nt], false, false);
        }
        __syncthreads();
    }

    // ---- epilogue (C/D layout: VGPR r -> M = r + 8*half, N = lane%16) ----
    const int mb = m0 + wave * 16 + ((lane >> 4) << 3);
#pragma unroll
    for (int nt = 0; nt < 8; ++nt) {
        int n = n0 + nt * 16 + (lane & 15);
        if (n >= N) continue;
        float bsv = (mode == 1) ? bias[n] : 0.f;
#pragma unroll
        for (int r = 0; r < 8; ++r) {
            int mm = mb + r;
            float v = acc[nt][r];
            if (mode == 1)      v = softplus_f(v + bsv);
            else if (mode == 2) v += resid[(size_t)mm * ldr + n];
            C[(size_t)mm * ldc + n] = v;
            if (Cbf) Cbf[(size_t)mm * ldc + n] = f2bf(v);
        }
    }
}

// ---------------------------------------------------------------- causal depthwise conv + SiLU
__global__ void __launch_bounds__(256)
conv_silu_kernel(const float* __restrict__ xz, const float* __restrict__ cw,
                 const float* __restrict__ cb, float* __restrict__ xb,
                 __bf16* __restrict__ xbbf) {
    int idx = blockIdx.x * 256 + threadIdx.x;          // over B*L*D_INNER
    if (idx >= BATCH * SEQLEN * D_INNER) return;
    int d = idx % D_INNER;
    int t = (idx / D_INNER) % SEQLEN;
    int b = idx / (D_INNER * SEQLEN);
    float acc = cb[d];
#pragma unroll
    for (int k = 0; k < D_CONV; ++k) {
        int tt = t - (D_CONV - 1) + k;
        if (tt >= 0)
            acc += xz[((size_t)(b * SEQLEN + tt)) * (2 * D_INNER) + d] * cw[d * D_CONV + k];
    }
    float s = silu_f(acc);
    xb[idx]   = s;                                     // row = b*L+t, stride D_INNER
    xbbf[idx] = f2bf(s);
}

// ---------------------------------------------------------------- selective scan
// lane = (channel, state); 512 threads = 32 channels x 16 states per block
__global__ void __launch_bounds__(512)
scan_kernel(const float* __restrict__ delta, const float* __restrict__ xb,
            const float* __restrict__ xpj, const float* __restrict__ xz,
            const float* __restrict__ A_log, const float* __restrict__ Dp,
            __bf16* __restrict__ ybf) {
    const int tid = threadIdx.x;
    const int n  = tid & 15;
    const int dl = tid >> 4;                       // 0..31
    const int cb = blockIdx.x % (D_INNER / 32);
    const int b  = blockIdx.x / (D_INNER / 32);
    const int d  = cb * 32 + dl;

    const float a   = -__expf(A_log[d * D_STATE + n]);
    const float dpv = Dp[d];
    float h = 0.f;
    const size_t rowbase = (size_t)b * SEQLEN;

    for (int t = 0; t < SEQLEN; ++t) {
        const size_t row = rowbase + t;
        float dv = delta[row * D_INNER + d];
        float xv = xb[row * D_INNER + d];
        float Bv = xpj[row * XPJ_N + DT_RANK + n];
        float Cv = xpj[row * XPJ_N + DT_RANK + D_STATE + n];
        h = __expf(dv * a) * h + dv * Bv * xv;
        float y = h * Cv;
        y += __shfl_xor(y, 1, 16);
        y += __shfl_xor(y, 2, 16);
        y += __shfl_xor(y, 4, 16);
        y += __shfl_xor(y, 8, 16);
        if (n == 0) {
            float yo = y + xv * dpv;
            float z  = xz[row * (2 * D_INNER) + D_INNER + d];
            ybf[row * D_INNER + d] = f2bf(yo * silu_f(z));
        }
    }
}

// ---------------------------------------------------------------- launch
extern "C" void kernel_launch(void* const* d_in, const int* in_sizes, int n_in,
                              void* d_out, int out_size, void* d_ws, size_t ws_size,
                              hipStream_t stream) {
    const float* x         = (const float*)d_in[0];
    const float* ln_gamma  = (const float*)d_in[1];
    const float* ln_beta   = (const float*)d_in[2];
    const float* in_proj_w = (const float*)d_in[3];   // 1024 x 4096
    const float* conv_w    = (const float*)d_in[4];   // 2048 x 4
    const float* conv_b    = (const float*)d_in[5];
    const float* x_proj_w  = (const float*)d_in[6];   // 2048 x 160
    const float* dt_proj_w = (const float*)d_in[7];   // 128 x 2048
    const float* dt_proj_b = (const float*)d_in[8];
    const float* A_log     = (const float*)d_in[9];   // 2048 x 16
    const float* D_param   = (const float*)d_in[10];
    const float* out_proj_w= (const float*)d_in[11];  // 2048 x 1024
    float* out = (float*)d_out;

    // carve workspace
    size_t off = 0;
    auto carve = [&](size_t bytes) -> char* {
        char* p = (char*)d_ws + off;
        off += (bytes + 255) & ~(size_t)255;
        return p;
    };
    __bf16* h_bf     = (__bf16*)carve((size_t)NROWS * D_MODEL * 2);
    __bf16* wt_in_bf = (__bf16*)carve((size_t)D_MODEL * 2 * D_INNER * 2);  // (4096 x 1024)
    float*  xz       = (float*) carve((size_t)NROWS * 2 * D_INNER * 4);
    float*  xb       = (float*) carve((size_t)NROWS * D_INNER * 4);
    __bf16* xb_bf    = (__bf16*)carve((size_t)NROWS * D_INNER * 2);
    __bf16* wt_xp_bf = (__bf16*)carve((size_t)D_INNER * XPJ_N * 2);        // (160 x 2048)
    float*  xpj      = (float*) carve((size_t)NROWS * XPJ_N * 4);
    __bf16* xpj_bf   = (__bf16*)carve((size_t)NROWS * XPJ_N * 2);
    __bf16* wt_dt_bf = (__bf16*)carve((size_t)DT_RANK * D_INNER * 2);      // (2048 x 128)
    float*  delta    = (float*) carve((size_t)NROWS * D_INNER * 4);
    __bf16* y_bf     = (__bf16*)carve((size_t)NROWS * D_INNER * 2);
    __bf16* wt_out_bf= (__bf16*)carve((size_t)D_INNER * D_MODEL * 2);      // (1024 x 2048)
    (void)ws_size; (void)n_in; (void)in_sizes; (void)out_size;

    // 1) layernorm + bf16 cast
    ln_cast_kernel<<<NROWS, 256, 0, stream>>>(x, ln_gamma, ln_beta, h_bf);

    // 2) weight transpose-casts (one-time; makes GEMM B-tile staging pure b128)
    {
        int n1 = D_MODEL * 2 * D_INNER;
        transpose_cast_bf16_kernel<<<(n1 + 255) / 256, 256, 0, stream>>>(
            in_proj_w, wt_in_bf, D_MODEL, 2 * D_INNER);
        int n2 = D_INNER * XPJ_N;
        transpose_cast_bf16_kernel<<<(n2 + 255) / 256, 256, 0, stream>>>(
            x_proj_w, wt_xp_bf, D_INNER, XPJ_N);
        int n3 = DT_RANK * D_INNER;
        transpose_cast_bf16_kernel<<<(n3 + 255) / 256, 256, 0, stream>>>(
            dt_proj_w, wt_dt_bf, DT_RANK, D_INNER);
        int n4 = D_INNER * D_MODEL;
        transpose_cast_bf16_kernel<<<(n4 + 255) / 256, 256, 0, stream>>>(
            out_proj_w, wt_out_bf, D_INNER, D_MODEL);
    }

    // 3) in_proj: xz = h @ in_proj_w   (2048 x 1024 x 4096)
    gemm_bf16_wmma<<<dim3((2 * D_INNER + TN - 1) / TN, NROWS / TM), 128, 0, stream>>>(
        h_bf, D_MODEL, wt_in_bf, xz, 2 * D_INNER,
        nullptr, nullptr, nullptr, 0, NROWS, 2 * D_INNER, D_MODEL, 0);

    // 4) causal depthwise conv + SiLU
    {
        int n = BATCH * SEQLEN * D_INNER;
        conv_silu_kernel<<<(n + 255) / 256, 256, 0, stream>>>(xz, conv_w, conv_b, xb, xb_bf);
    }

    // 5) x_proj: xpj = xb @ x_proj_w   (2048 x 2048 x 160), also bf16 copy for dt GEMM
    gemm_bf16_wmma<<<dim3((XPJ_N + TN - 1) / TN, NROWS / TM), 128, 0, stream>>>(
        xb_bf, D_INNER, wt_xp_bf, xpj, XPJ_N,
        xpj_bf, nullptr, nullptr, 0, NROWS, XPJ_N, D_INNER, 0);

    // 6) delta = softplus(dt_r @ dt_proj_w + dt_proj_b)   (2048 x 128 x 2048)
    gemm_bf16_wmma<<<dim3(D_INNER / TN, NROWS / TM), 128, 0, stream>>>(
        xpj_bf, XPJ_N, wt_dt_bf, delta, D_INNER,
        nullptr, dt_proj_b, nullptr, 0, NROWS, D_INNER, DT_RANK, 1);

    // 7) selective scan (+ D term, SiLU(z) gate, bf16 cast)
    scan_kernel<<<BATCH * (D_INNER / 32), 512, 0, stream>>>(
        delta, xb, xpj, xz, A_log, D_param, y_bf);

    // 8) out_proj + residual: out = y @ out_proj_w + x   (2048 x 2048 x 1024)
    gemm_bf16_wmma<<<dim3(D_MODEL / TN, NROWS / TM), 128, 0, stream>>>(
        y_bf, D_INNER, wt_out_bf, out, D_MODEL,
        nullptr, nullptr, x, D_MODEL, NROWS, D_MODEL, D_INNER, 2);
}
